// JumpDotProduct_22179211116606
// MI455X (gfx1250) — compile-verified, matrix operand order checked by reading, not run on hardware
//
#include <hip/hip_runtime.h>

#define BATCH  8
#define SEQ    400
#define SDIM   512
#define DMODEL 256

typedef __attribute__((ext_vector_type(16))) _Float16 v16h;
typedef __attribute__((ext_vector_type(8)))  _Float16 v8h;
typedef __attribute__((ext_vector_type(8)))  float    v8f;
typedef __attribute__((ext_vector_type(4)))  float    v4f;

// A/B fragment for v_wmma_f32_16x16x32_f16.
// Lane L: row/col index = L%16, K-half = L/16.
// Per ISA layout: elems 0..7 = K[kb + khalf*8 .. +7], elems 8..15 = K[kb+16+khalf*8 .. +7]
static __device__ __forceinline__ v16h frag_from_f16(const _Float16* lanebase, int kb, int khalf) {
  const v8h* p0 = reinterpret_cast<const v8h*>(lanebase + kb + khalf * 8);
  const v8h* p1 = reinterpret_cast<const v8h*>(lanebase + kb + 16 + khalf * 8);
  v8h lo = *p0, hi = *p1;
  v16h r;
#pragma unroll
  for (int t = 0; t < 8; ++t) { r[t] = lo[t]; r[t + 8] = hi[t]; }
  return r;
}

// Same fragment but sourced from f32 memory (rel_sinusoid stream), converted in-register.
static __device__ __forceinline__ v16h frag_from_f32(const float* lanebase, int kb, int khalf) {
  const v4f* p0 = reinterpret_cast<const v4f*>(lanebase + kb + khalf * 8);
  const v4f* p1 = reinterpret_cast<const v4f*>(lanebase + kb + 16 + khalf * 8);
  v4f a0 = p0[0], a1 = p0[1], b0 = p1[0], b1 = p1[1];
  v16h r;
#pragma unroll
  for (int t = 0; t < 4; ++t) {
    r[t]      = (_Float16)a0[t];
    r[t + 4]  = (_Float16)a1[t];
    r[t + 8]  = (_Float16)b0[t];
    r[t + 12] = (_Float16)b1[t];
  }
  return r;
}

// ---------------- Stage 0 kernels ----------------

__global__ void cvt16(const float* __restrict__ src, _Float16* __restrict__ dst, int n) {
  int i = blockIdx.x * blockDim.x + threadIdx.x;
  if (i < n) dst[i] = (_Float16)src[i];
}

// wt[c][r] = (f16) w[r][c]  (transpose-convert: B-operand wants [n][k] with K contiguous)
__global__ void wcvt_t(const float* __restrict__ w, _Float16* __restrict__ wt, int R, int C) {
  int idx = blockIdx.x * blockDim.x + threadIdx.x;
  if (idx >= R * C) return;
  int r = idx / C, c = idx % C;
  wt[(size_t)c * R + r] = (_Float16)w[idx];
}

// wqrT[e][c] = f16( sum_d w_q2[c,d] * w_k2[e,d] )   (folds w_k2 into the q2 projection)
__global__ void wqr_dot(const float* __restrict__ wq2, const float* __restrict__ wk2,
                        _Float16* __restrict__ wqrT) {
  int idx = blockIdx.x * blockDim.x + threadIdx.x;
  if (idx >= SDIM * DMODEL) return;
  int c = idx / DMODEL, e = idx % DMODEL;
  const float* a = wq2 + (size_t)c * DMODEL;
  const float* b = wk2 + (size_t)e * DMODEL;
  float s = 0.f;
#pragma unroll 8
  for (int d = 0; d < DMODEL; ++d) s = fmaf(a[d], b[d], s);
  wqrT[(size_t)e * SDIM + c] = (_Float16)s;
}

// ---------------- Stage 1: f16 WMMA GEMM (one 16x16 tile per wave) ----------------
// C[M x N] (f16) = A[M x K] (f16, row major) @ BT[N x K]^T (f16, row major)
__global__ void gemm_xw_f16(const _Float16* __restrict__ A, const _Float16* __restrict__ BT,
                            _Float16* __restrict__ C, int M, int N, int K) {
  int wave = blockIdx.x * (blockDim.x >> 5) + (threadIdx.x >> 5);
  int ntiles = (M >> 4) * (N >> 4);
  if (wave >= ntiles) return;  // wave-uniform guard
  int lane = threadIdx.x & 31;
  int it = wave / (N >> 4);
  int jt = wave % (N >> 4);
  int r16 = lane & 15, khalf = lane >> 4;
  const _Float16* abase = A + (size_t)(it * 16 + r16) * K;
  const _Float16* bbase = BT + (size_t)(jt * 16 + r16) * K;
  v8f acc = {};
  for (int kb = 0; kb < K; kb += 32) {
    v16h af = frag_from_f16(abase, kb, khalf);
    v16h bf = frag_from_f16(bbase, kb, khalf);
    acc = __builtin_amdgcn_wmma_f32_16x16x32_f16(false, af, false, bf, (short)0, acc, false, false);
  }
  // D layout: VGPR r, lanes 0-15 -> m=r, lanes 16-31 -> m=r+8; n = lane%16
  _Float16* cptr = C + (size_t)(it * 16) * N + jt * 16 + r16;
#pragma unroll
  for (int r = 0; r < 8; ++r)
    cptr[(size_t)(r + 8 * khalf) * N] = (_Float16)acc[r];
}

// ---------------- Stage 2a: edge = q @ k^T / 16 ----------------
__global__ void edge_qk(const _Float16* __restrict__ q16, const _Float16* __restrict__ k16,
                        float* __restrict__ out) {
  int wave = blockIdx.x * (blockDim.x >> 5) + (threadIdx.x >> 5);
  if (wave >= BATCH * 25 * 25) return;
  int lane = threadIdx.x & 31;
  int b = wave / 625, rem = wave % 625;
  int it = rem / 25, jt = rem % 25;
  int r16 = lane & 15, khalf = lane >> 4;
  const _Float16* abase = q16 + ((size_t)b * SEQ + it * 16 + r16) * DMODEL;
  const _Float16* bbase = k16 + ((size_t)b * SEQ + jt * 16 + r16) * DMODEL;
  v8f acc = {};
#pragma unroll
  for (int kb = 0; kb < DMODEL; kb += 32) {
    v16h af = frag_from_f16(abase, kb, khalf);
    v16h bf = frag_from_f16(bbase, kb, khalf);
    acc = __builtin_amdgcn_wmma_f32_16x16x32_f16(false, af, false, bf, (short)0, acc, false, false);
  }
  float* optr = out + (size_t)b * SEQ * SEQ + (size_t)(it * 16) * SEQ + jt * 16 + r16;
#pragma unroll
  for (int r = 0; r < 8; ++r)
    optr[(size_t)(r + 8 * khalf) * SEQ] = acc[r] * 0.0625f;
}

// ---------------- Stage 2b: out[b,i,j] += (qr[b,i,:] . rel_sin[i,j,:]) / 16 ----------------
// Tile: fixed i, M rows 0..7 = batches (8..15 zero), N = 16 j's. Streams rel_sin (f32) once.
__global__ void rel_add(const _Float16* __restrict__ qr16, const float* __restrict__ rel,
                        float* __restrict__ out) {
  int wave = blockIdx.x * (blockDim.x >> 5) + (threadIdx.x >> 5);
  if (wave >= SEQ * 25) return;
  int lane = threadIdx.x & 31;
  int i = wave / 25, jt = wave % 25;
  int r16 = lane & 15, khalf = lane >> 4;
  const _Float16* abase = qr16 + ((size_t)(r16 & 7) * SEQ + i) * DMODEL;  // valid addr even for r16>=8
  const float* bbase = rel + ((size_t)i * SEQ + jt * 16 + r16) * DMODEL;
  v8f acc = {};
#pragma unroll
  for (int kb = 0; kb < DMODEL; kb += 32) {
    v16h af = {};
    if (r16 < 8) af = frag_from_f16(abase, kb, khalf);  // rows 8..15 stay zero
    if (kb + 32 < DMODEL) __builtin_prefetch(bbase + kb + 32, 0, 0);
    v16h bf = frag_from_f32(bbase, kb, khalf);
    acc = __builtin_amdgcn_wmma_f32_16x16x32_f16(false, af, false, bf, (short)0, acc, false, false);
  }
  if (khalf == 0) {  // m=0..7 live in VGPRs 0..7 of lanes 0..15
    float* optr = out + (size_t)i * SEQ + jt * 16 + r16;
#pragma unroll
    for (int r = 0; r < 8; ++r)
      optr[(size_t)r * SEQ * SEQ] += acc[r] * 0.0625f;
  }
}

// ---------------- Stage 2c: alignment mask (int32 ones outer product) ----------------
__global__ void mask_outer(const int* __restrict__ mask, int* __restrict__ om) {
  int t = blockIdx.x * blockDim.x + threadIdx.x;
  if (t >= BATCH * SEQ * SEQ) return;
  int b = t / (SEQ * SEQ), r = t % (SEQ * SEQ);
  int i = r / SEQ, j = r % SEQ;
  om[t] = mask[b * SEQ + i] * mask[b * SEQ + j];
}

extern "C" void kernel_launch(void* const* d_in, const int* in_sizes, int n_in,
                              void* d_out, int out_size, void* d_ws, size_t ws_size,
                              hipStream_t stream) {
  const float* x    = (const float*)d_in[0];
  const int*   mask = (const int*)d_in[1];
  const float* w_q  = (const float*)d_in[2];
  const float* w_q2 = (const float*)d_in[3];
  const float* w_k  = (const float*)d_in[4];
  const float* w_k2 = (const float*)d_in[5];
  const float* rel  = (const float*)d_in[6];

  char* ws = (char*)d_ws;
  _Float16* x16   = (_Float16*)(ws + 0);        // 3,276,800 B
  _Float16* wq16T = (_Float16*)(ws + 3276800);  //   262,144 B
  _Float16* wk16T = (_Float16*)(ws + 3538944);  //   262,144 B
  _Float16* wqrT  = (_Float16*)(ws + 3801088);  //   262,144 B
  _Float16* q16   = (_Float16*)(ws + 4063232);  // 1,638,400 B
  _Float16* k16   = (_Float16*)(ws + 5701632);  // 1,638,400 B
  _Float16* qr16  = (_Float16*)(ws + 7340032);  // 1,638,400 B  (total ~8.98 MB)

  float* edge = (float*)d_out;
  int* amask = (int*)d_out + (size_t)BATCH * SEQ * SEQ;

  // Stage 0: converts + combined weight
  cvt16<<<(BATCH * SEQ * SDIM + 255) / 256, 256, 0, stream>>>(x, x16, BATCH * SEQ * SDIM);
  wcvt_t<<<(SDIM * DMODEL + 255) / 256, 256, 0, stream>>>(w_q, wq16T, SDIM, DMODEL);
  wcvt_t<<<(SDIM * DMODEL + 255) / 256, 256, 0, stream>>>(w_k, wk16T, SDIM, DMODEL);
  wqr_dot<<<(SDIM * DMODEL + 255) / 256, 256, 0, stream>>>(w_q2, w_k2, wqrT);

  // Stage 1: projections q16/k16/qr16 = x16 @ W  (M=3200, N=256, K=512; 3200 tiles, 8 waves/block)
  gemm_xw_f16<<<400, 256, 0, stream>>>(x16, wq16T, q16, 3200, DMODEL, SDIM);
  gemm_xw_f16<<<400, 256, 0, stream>>>(x16, wk16T, k16, 3200, DMODEL, SDIM);
  gemm_xw_f16<<<400, 256, 0, stream>>>(x16, wqrT, qr16, 3200, DMODEL, SDIM);

  // Stage 2
  edge_qk<<<625, 256, 0, stream>>>(q16, k16, edge);        // 5000 tiles
  rel_add<<<1250, 256, 0, stream>>>(qr16, rel, edge);      // 10000 tiles, streams 164MB once
  mask_outer<<<(BATCH * SEQ * SEQ + 255) / 256, 256, 0, stream>>>(mask, amask);
}
